// RelationRouterMoE_47768626266205
// MI455X (gfx1250) — matrix operand order repaired and checked by reading, not compiled, not run on hardware
//
#include <hip/hip_runtime.h>

// ---------------------------------------------------------------------------
// Problem constants (from reference)
// ---------------------------------------------------------------------------
static constexpr int B_   = 4;
static constexpr int R_   = 256;
static constexpr int E_   = 8192;
static constexpr int M_   = 16;
static constexpr int NEXP_ = 8;
static constexpr int CAP_  = 1280;   // ceil(1.25*E/NEXP)
static constexpr int BUD_  = 7864;   // int(0.12*R*R)
static constexpr float TAU_ = 2.0f;

// Edge tile per block: 32 edges = 2 WMMA M-tiles.
static constexpr int ET_ = 32;

// TDM availability (arity differs across toolchains)
#if defined(__has_builtin)
#  if __has_builtin(__builtin_amdgcn_tensor_load_to_lds)
#    define HAVE_TDM 1
#  endif
#endif
#ifndef HAVE_TDM
#  define HAVE_TDM 0
#endif

// ---------------------------------------------------------------------------
// WMMA types / helpers (CDNA5 wave32, V_WMMA_F32_16X16X32_BF16)
// ---------------------------------------------------------------------------
typedef __attribute__((ext_vector_type(16))) __bf16    v16bf;
typedef __attribute__((ext_vector_type(8)))  float     v8f;
typedef __attribute__((ext_vector_type(8)))  unsigned  v8u;
typedef __attribute__((ext_vector_type(4)))  unsigned  u32x4;
typedef __attribute__((ext_vector_type(8)))  int       i32x8;
typedef __attribute__((ext_vector_type(4)))  int       i32x4;

__device__ __forceinline__ v16bf as_bf16x16(v8u u) {
  union { v8u u; v16bf b; } x; x.u = u; return x.b;
}

__device__ __forceinline__ v8f wmma_bf16(v8u a, v8u b, v8f c) {
  return __builtin_amdgcn_wmma_f32_16x16x32_bf16(
      false, as_bf16x16(a), false, as_bf16x16(b), (short)0, c, false, false);
}

__device__ __forceinline__ unsigned short f2bf(float f) {
  unsigned u = __float_as_uint(f);
  return (unsigned short)((u + 0x7FFFu + ((u >> 16) & 1u)) >> 16);   // RNE
}
__device__ __forceinline__ unsigned pack_bf2(float lo, float hi) {
  return (unsigned)f2bf(lo) | ((unsigned)f2bf(hi) << 16);
}
__device__ __forceinline__ float gelu_exact(float x) {
  return 0.5f * x * (1.0f + erff(x * 0.70710678118654752f));
}

// A fragment: 16x32 bf16 tile from LDS (row stride = 128 packed words).
// lanes 0-15 -> row=lane, K words {0..3, 8..11}+wbase;
// lanes 16-31 -> row=lane-16, K words {4..7, 12..15}+wbase.
__device__ __forceinline__ v8u load_a_frag(const unsigned* sA, int wbase) {
  int lane = threadIdx.x & 31;
  int half = lane >> 4, row = lane & 15;
  const unsigned* p = sA + row * 128 + wbase + half * 4;
  v8u u;
  u[0] = p[0]; u[1] = p[1]; u[2] = p[2]; u[3] = p[3];
  u[4] = p[8]; u[5] = p[9]; u[6] = p[10]; u[7] = p[11];
  return u;
}

// B fragment from *fragment-packed* storage: per (kt,nt) each lane's 8 words
// are contiguous (32B/lane) -> 2x b128 loads, fully coalesced.
__device__ __forceinline__ v8u load_b_frag_pk(const unsigned* W, int kt, int nt) {
  int lane = threadIdx.x & 31;
  const unsigned* p = W + ((((size_t)kt * 16 + nt) * 32 + lane) << 3);
  v8u u;
#pragma unroll
  for (int v = 0; v < 8; v++) u[v] = p[v];
  return u;
}

// simple stateless uniform RNG (JAX threefry not reproduced; no runtime check)
__device__ __forceinline__ float rng_uniform(unsigned x) {
  x ^= x >> 17; x *= 0xed5ad4bbu; x ^= x >> 11;
  x *= 0xac4c1b51u; x ^= x >> 15; x *= 0x31848babu; x ^= x >> 14;
  float u = (float)(x >> 8) * (1.0f / 16777216.0f);
  return fminf(fmaxf(u, 1e-6f), 1.0f - 1e-6f);
}

// ---------------------------------------------------------------------------
// TDM: stage nwords (multiple of 2) of packed weights into LDS as a 1-D tile
// of 8-byte elements. D# built per CDNA5 ISA 8.3/8.4 (type=2, count=1).
// ---------------------------------------------------------------------------
__device__ __forceinline__ void stage_weights(const unsigned* __restrict__ g,
                                              unsigned* sW, unsigned nwords) {
#if HAVE_TDM
  if ((threadIdx.x >> 5) == 0) {
    unsigned long long ga = (unsigned long long)(const void*)g;
    unsigned lds = (unsigned)(unsigned long long)(void*)sW;  // low 32b = LDS offset
    unsigned ne = nwords >> 1;                               // 8-byte elements
    u32x4 g0;
    g0[0] = 1u;                                              // count=1 (valid)
    g0[1] = lds;                                             // lds_addr
    g0[2] = (unsigned)(ga & 0xFFFFFFFFu);                    // global_addr lo
    g0[3] = (unsigned)((ga >> 32) & 0x01FFFFFFu) | (2u << 30); // hi | type=2
    i32x8 g1;
    g1[0] = (int)(3u << 16);                    // wg_mask=0, data_size=8B
    g1[1] = (int)((ne & 0xFFFFu) << 16);        // tensor_dim0[15:0]
    g1[2] = (int)((ne >> 16) & 0xFFFFu) | (1 << 16); // tdim0[31:16], tdim1=1
    g1[3] = (int)((ne & 0xFFFFu) << 16);        // tdim1 hi=0, tile_dim0=ne
    g1[4] = 1;                                  // tile_dim1=1, tile_dim2=0
    g1[5] = (int)ne;                            // tensor_dim0_stride lo
    g1[6] = 0;                                  // stride0 hi, stride1 lo
    g1[7] = 0;
    i32x4 z4 = {0, 0, 0, 0};
#if defined(__clang_major__) && (__clang_major__ >= 23)
    i32x8 z8 = {0, 0, 0, 0, 0, 0, 0, 0};
    __builtin_amdgcn_tensor_load_to_lds(g0, g1, z4, z4, z8, 0);
#else
    __builtin_amdgcn_tensor_load_to_lds(g0, g1, z4, z4, 0);
#endif
    __builtin_amdgcn_s_wait_tensorcnt(0);
  }
  __syncthreads();
#else
  for (unsigned i = threadIdx.x; i < nwords; i += blockDim.x) sW[i] = g[i];
  __syncthreads();
#endif
}

// ---------------------------------------------------------------------------
// Weight pre-pack: fp32 (Ksrc x 256) row-major -> fragment-packed bf16 pairs.
// Word index i = ((kt*16 + nt)*32 + lane)*8 + v holds K pair
// (kt*32 + (lane>>4)*8 + 2v + (v>=4 ? 8 : 0), +1) at column nt*16 + (lane&15).
// ---------------------------------------------------------------------------
__global__ void k_pack_weight_frag(const float* __restrict__ W, unsigned* __restrict__ dst,
                                   int Ksrc, int Ktiles) {
  const int N = 256;
  int i = blockIdx.x * blockDim.x + threadIdx.x;
  int total = Ktiles * 16 * 32 * 8;
  if (i >= total) return;
  int v = i & 7;
  int lane = (i >> 3) & 31;
  int nt = (i >> 8) & 15;
  int kt = i >> 12;
  int n = nt * 16 + (lane & 15);
  int k = kt * 32 + (lane >> 4) * 8 + 2 * v + (v >= 4 ? 8 : 0);
  float lo = (k < Ksrc)     ? W[(size_t)k * N + n]       : 0.f;
  float hi = (k + 1 < Ksrc) ? W[(size_t)(k + 1) * N + n] : 0.f;
  dst[i] = pack_bf2(lo, hi);
}

// fp32 activations (rows x 256) -> packed bf16 (rows x 128 words)
__global__ void k_pack_act(const float* __restrict__ src, unsigned* __restrict__ dst,
                           int nwords) {
  int i = blockIdx.x * blockDim.x + threadIdx.x;
  if (i >= nwords) return;
  dst[i] = pack_bf2(src[2 * i], src[2 * i + 1]);
}

// ---------------------------------------------------------------------------
// Generic row-tile WMMA GEMM: Y = X(rows x K) @ W(K x 256) + bias (fp32 out).
// B fragments direct from fragment-packed global (2x b128 per frag).
// ---------------------------------------------------------------------------
__global__ __launch_bounds__(256) void k_gemm_rows(
    const float* __restrict__ X, int K,
    const unsigned* __restrict__ Wpk, const float* __restrict__ bias,
    float* __restrict__ Y) {
  __shared__ unsigned sA[16 * 128];
  const int rows0 = blockIdx.x * 16;
  const int kw = K / 2, ktiles = K / 32;
  for (int i = threadIdx.x; i < 16 * kw; i += 256) {
    int r = i / kw, w = i - r * kw;
    const float* x = X + (size_t)(rows0 + r) * K + 2 * w;
    sA[r * 128 + w] = pack_bf2(x[0], x[1]);
  }
  __syncthreads();
  const int lane = threadIdx.x & 31, wave = threadIdx.x >> 5;
  const int half = lane >> 4;
  for (int t = wave; t < 16; t += 8) {
    v8f acc = {};
    for (int kt = 0; kt < ktiles; kt++)
      acc = wmma_bf16(load_a_frag(sA, kt * 16), load_b_frag_pk(Wpk, kt, t), acc);
    int col = t * 16 + (lane & 15);
#pragma unroll
    for (int r = 0; r < 8; r++)
      Y[(size_t)(rows0 + r + half * 8) * 256 + col] = acc[r] + bias[col];
  }
}

// ---------------------------------------------------------------------------
// LN on (ET_ x 256) fp32 LDS tile -> packed bf16. One wave per row group;
// each lane owns 8 contiguous cols so bf16 pairs stay in-lane.
// ---------------------------------------------------------------------------
__device__ __forceinline__ void ln_rows_to_bf16(const float* sF, unsigned* sA,
                                                const float* __restrict__ g,
                                                const float* __restrict__ be) {
  const int lane = threadIdx.x & 31, wave = threadIdx.x >> 5;
  for (int rr = wave; rr < ET_; rr += 8) {
    float v[8], s = 0.f, s2 = 0.f;
#pragma unroll
    for (int j = 0; j < 8; j++) {
      float x = sF[rr * 256 + lane * 8 + j];
      v[j] = x; s += x; s2 += x * x;
    }
#pragma unroll
    for (int off = 16; off >= 1; off >>= 1) {
      s  += __shfl_xor(s,  off, 32);
      s2 += __shfl_xor(s2, off, 32);
    }
    float mu  = s * (1.f / 256.f);
    float var = s2 * (1.f / 256.f) - mu * mu;
    float inv = rsqrtf(var + 1e-5f);
#pragma unroll
    for (int j = 0; j < 4; j++) {
      int c = lane * 8 + 2 * j;
      float lo = (v[2 * j]     - mu) * inv * g[c]     + be[c];
      float hi = (v[2 * j + 1] - mu) * inv * g[c + 1] + be[c + 1];
      sA[rr * 128 + (c >> 1)] = pack_bf2(lo, hi);
    }
  }
}

// GEMM over 2 M-tiles x (2 N-tiles per wave), B served from LDS slab.
// epi(nt, mt, acc) applies the stage epilogue.
template <typename F>
__device__ __forceinline__ void gemm_tiles(const unsigned* sW, const unsigned* sA,
                                           int ktiles, F&& epi) {
  const int wave = threadIdx.x >> 5;
#pragma unroll
  for (int q = 0; q < 2; q++) {
    int nt = wave * 2 + q;
    v8f a0 = {}, a1 = {};
    for (int kt = 0; kt < ktiles; kt++) {
      v8u b = load_b_frag_pk(sW, kt, nt);
      a0 = wmma_bf16(load_a_frag(sA, kt * 16), b, a0);
      a1 = wmma_bf16(load_a_frag(sA + 16 * 128, kt * 16), b, a1);
    }
    epi(nt, 0, a0);
    epi(nt, 1, a1);
  }
}

// ---------------------------------------------------------------------------
// Fused per-edge pipeline. Block = 32 edges (2 M-tiles), 256 thr (8 waves).
// Weight slabs staged into LDS via TDM; B frags from LDS (ds_load_b128).
// Dynamic LDS: 32768(sW) + 8192(sF) + 4*4096(sA..sP3) + 64 + 256 words.
// ---------------------------------------------------------------------------
static constexpr unsigned SMEM_WORDS_ = 32768u + 8192u + 4u * 4096u + 64u + 256u;

__global__ __launch_bounds__(256) void k_edges(
    const float* __restrict__ edge_bank, const int* __restrict__ edge_index,
    const unsigned* __restrict__ node_bfpk, const float* __restrict__ comm,
    const unsigned* __restrict__ WedgePk, const float* __restrict__ b_edge,
    const float* __restrict__ ln_c_g, const float* __restrict__ ln_c_b,
    const unsigned* __restrict__ WcluPk, const float* __restrict__ b_clu,
    const unsigned* __restrict__ WtokPk, const float* __restrict__ b_tok,
    const unsigned* __restrict__ Wr1Pk, const float* __restrict__ b_r1,
    const float* __restrict__ ln_r_g, const float* __restrict__ ln_r_b,
    const unsigned* __restrict__ Wr2Pk, const float* __restrict__ b_r2,
    const float* __restrict__ head_prompts,
    int* __restrict__ relW, float* __restrict__ scoreW, float* __restrict__ accums) {
  extern __shared__ unsigned smem[];
  unsigned* sW  = smem;                           // 32768 words (128KB slab)
  float*    sF  = (float*)(smem + 32768);         // 32x256 fp32
  unsigned* sA  = smem + 32768 + 8192;            // 32x128 packed bf16
  unsigned* sP1 = sA + 4096;
  unsigned* sP2 = sP1 + 4096;
  unsigned* sP3 = sP2 + 4096;
  int*      sSrc = (int*)(sP3 + 4096);
  int*      sDst = sSrc + ET_;
  float*    sLog = (float*)(sDst + ET_);          // 32x8

  const int tid  = threadIdx.x;
  const int lane = tid & 31, half = lane >> 4;
  const int b  = blockIdx.x / (E_ / ET_);
  const int e0 = (blockIdx.x % (E_ / ET_)) * ET_;

  __builtin_prefetch(WtokPk, 0, 3);
  __builtin_prefetch(Wr1Pk, 0, 3);
  __builtin_prefetch(Wr2Pk, 0, 3);

  if (tid < ET_) {
    sSrc[tid] = edge_index[(size_t)(b * 2 + 0) * E_ + e0 + tid];
    sDst[tid] = edge_index[(size_t)(b * 2 + 1) * E_ + e0 + tid];
  }
  // stage edge_bank tile (ET_ x 16, K padded to 32)
  for (int i = tid; i < ET_ * 16; i += 256) {
    int r = i >> 4, w = i & 15;
    unsigned v = 0u;
    if (w < 8) {
      const float* x = edge_bank + (size_t)(b * E_ + e0 + r) * M_ + 2 * w;
      v = pack_bf2(x[0], x[1]);
    }
    sA[r * 128 + w] = v;
  }
  __syncthreads();

  // edge_feat = bank @ W_edge + b_edge
  stage_weights(WedgePk, sW, 16 * 256);
  gemm_tiles(sW, sA, 1, [&](int nt, int mt, v8f acc) {
    int col = nt * 16 + (lane & 15);
#pragma unroll
    for (int r = 0; r < 8; r++)
      sF[(mt * 16 + r + half * 8) * 256 + col] = acc[r] + b_edge[col];
  });
  __syncthreads();

  ln_rows_to_bf16(sF, sA, ln_c_g, ln_c_b);
  __syncthreads();

  // edge_feat += gelu(LN @ W_clu + b_clu)
  stage_weights(WcluPk, sW, 128 * 256);
  gemm_tiles(sW, sA, 8, [&](int nt, int mt, v8f acc) {
    int col = nt * 16 + (lane & 15);
#pragma unroll
    for (int r = 0; r < 8; r++)
      sF[(mt * 16 + r + half * 8) * 256 + col] += gelu_exact(acc[r] + b_clu[col]);
  });
  __syncthreads();

  // token parts: p0 = bf16(edge_feat); p1/p2 gathered node feats; p3 comm |diff|
  for (int i = tid; i < ET_ * 128; i += 256) {
    int r = i >> 7, w = i & 127;
    sA[i] = pack_bf2(sF[r * 256 + 2 * w], sF[r * 256 + 2 * w + 1]);
    int s = sSrc[r], d = sDst[r];
    sP1[i] = node_bfpk[(size_t)(b * R_ + s) * 128 + w];
    sP2[i] = node_bfpk[(size_t)(b * R_ + d) * 128 + w];
    const float* cs = comm + (size_t)(b * R_ + s) * 256 + 2 * w;
    const float* cd = comm + (size_t)(b * R_ + d) * 256 + 2 * w;
    sP3[i] = pack_bf2(fabsf(cs[0] - cd[0]), fabsf(cs[1] - cd[1]));
  }
  __syncthreads();

  // token = concat(p0..p3, zeros) @ W_tok + b_tok (zeros slice dropped).
  // Accumulate across the 4 slabs; restage sW per slab.
  {
    const unsigned* parts[4] = {sA, sP1, sP2, sP3};
    const int wave = tid >> 5;
    v8f tacc[2][2] = {{{}, {}}, {{}, {}}};
    for (int p = 0; p < 4; p++) {
      stage_weights(WtokPk + (size_t)p * 128 * 256, sW, 128 * 256);
      const unsigned* part = parts[p];
#pragma unroll
      for (int q = 0; q < 2; q++) {
        int nt = wave * 2 + q;
        for (int kt = 0; kt < 8; kt++) {
          v8u bb = load_b_frag_pk(sW, kt, nt);
          tacc[q][0] = wmma_bf16(load_a_frag(part, kt * 16), bb, tacc[q][0]);
          tacc[q][1] = wmma_bf16(load_a_frag(part + 16 * 128, kt * 16), bb, tacc[q][1]);
        }
      }
      __syncthreads();
    }
#pragma unroll
    for (int q = 0; q < 2; q++) {
      int nt = wave * 2 + q, col = nt * 16 + (lane & 15);
#pragma unroll
      for (int mt = 0; mt < 2; mt++)
#pragma unroll
        for (int r = 0; r < 8; r++)
          sF[(mt * 16 + r + half * 8) * 256 + col] = tacc[q][mt][r] + b_tok[col];
    }
  }
  __syncthreads();
  for (int i = tid; i < ET_ * 128; i += 256) {
    int r = i >> 7, w = i & 127;
    sA[i] = pack_bf2(sF[r * 256 + 2 * w], sF[r * 256 + 2 * w + 1]);
  }
  __syncthreads();

  // h = gelu(token @ W_r1 + b_r1)
  stage_weights(Wr1Pk, sW, 128 * 256);
  gemm_tiles(sW, sA, 8, [&](int nt, int mt, v8f acc) {
    int col = nt * 16 + (lane & 15);
#pragma unroll
    for (int r = 0; r < 8; r++)
      sF[(mt * 16 + r + half * 8) * 256 + col] = gelu_exact(acc[r] + b_r1[col]);
  });
  __syncthreads();
  ln_rows_to_bf16(sF, sA, ln_r_g, ln_r_b);
  __syncthreads();

  // routed = gelu(LN(h) @ W_r2 + b_r2)
  stage_weights(Wr2Pk, sW, 128 * 256);
  gemm_tiles(sW, sA, 8, [&](int nt, int mt, v8f acc) {
    int col = nt * 16 + (lane & 15);
#pragma unroll
    for (int r = 0; r < 8; r++)
      sF[(mt * 16 + r + half * 8) * 256 + col] = gelu_exact(acc[r] + b_r2[col]);
  });
  __syncthreads();

  // logits: 32 edges x 8 experts = exactly 256 threads
  {
    int e = tid >> 3, m = tid & 7;
    const float* hp = head_prompts + (size_t)m * 256;
    float acc = 0.f;
    for (int k = 0; k < 256; k++) acc += sF[e * 256 + k] * hp[k];
    sLog[e * 8 + m] = acc;
  }
  __syncthreads();

  // Gumbel routing. probs_st == hard numerically -> score = 1.0,
  // relation_id = argmax(y) (top-2 mask cannot change the argmax).
  if (tid < ET_) {
    int e = e0 + tid;
    float z[8], mx = -1e30f;
#pragma unroll
    for (int m = 0; m < 8; m++) {
      float u = rng_uniform((unsigned)((b * E_ + e) * 8 + m) * 2654435761u + 42u);
      float g = -logf(-logf(u));
      z[m] = (sLog[tid * 8 + m] + g) / TAU_;
      mx = fmaxf(mx, z[m]);
    }
    float s = 0.f, y[8];
#pragma unroll
    for (int m = 0; m < 8; m++) { y[m] = expf(z[m] - mx); s += y[m]; }
    int i1 = 0;
#pragma unroll
    for (int m = 1; m < 8; m++) if (y[m] > y[i1]) i1 = m;  // ties -> lowest idx
    relW[(size_t)b * E_ + e]   = i1;
    scoreW[(size_t)b * E_ + e] = 1.0f;
    atomicAdd(&accums[i1], 1.0f);  // load histogram (hard one-hot mean)
  }
}

// ---------------------------------------------------------------------------
// capacity -> guard -> budget -> pack, one wave per batch (ballot scans;
// scores are uniform so stable ranks == order-preserving prefix counts).
// ---------------------------------------------------------------------------
__global__ __launch_bounds__(32) void k_select_pack(
    const int* __restrict__ edge_index, const int* __restrict__ relW,
    const float* __restrict__ scoreW, int* __restrict__ keepWS,
    float* __restrict__ out, float* __restrict__ accums) {
  __shared__ int cov[R_];
  __shared__ int bestBits[R_];
  const int b = blockIdx.x;
  const int lane = threadIdx.x;
  const unsigned ltmask = (1u << lane) - 1u;
  const int* rel = relW + (size_t)b * E_;
  const float* sc = scoreW + (size_t)b * E_;
  const int* srcA = edge_index + (size_t)(b * 2 + 0) * E_;
  const int* dstA = edge_index + (size_t)(b * 2 + 1) * E_;
  int* keep = keepWS + (size_t)b * E_;

  // capacity
  int cnt[NEXP_];
#pragma unroll
  for (int r = 0; r < NEXP_; r++) cnt[r] = 0;
  int capKept = 0;
  for (int base = 0; base < E_; base += 32) {
    int e = base + lane;
    int r = rel[e], k = 0;
#pragma unroll
    for (int rr = 0; rr < NEXP_; rr++) {
      bool is = (r == rr);
      unsigned m = (unsigned)__ballot(is);
      if (is) k = (cnt[rr] + __popc(m & ltmask)) < CAP_ ? 1 : 0;
      cnt[rr] += __popc(m);
    }
    keep[e] = k;
    capKept += k;
  }
#pragma unroll
  for (int off = 16; off >= 1; off >>= 1) capKept += __shfl_xor(capKept, off, 32);
  if (lane == 0) atomicAdd(&accums[8], (float)capKept);  // for overflow

  // guard
  for (int i = lane; i < R_; i += 32) { cov[i] = 0; bestBits[i] = 0; }
  __syncthreads();
  for (int base = 0; base < E_; base += 32) {
    int e = base + lane;
    int k = keep[e], si = srcA[e], di = dstA[e];
    int sb = __float_as_int(sc[e]);  // scores > 0 -> int-max == float-max
    if (k) { atomicMax(&cov[si], 1); atomicMax(&cov[di], 1); }
    atomicMax(&bestBits[si], sb);
    atomicMax(&bestBits[di], sb);
  }
  __syncthreads();
  for (int base = 0; base < E_; base += 32) {
    int e = base + lane;
    float s = sc[e];
    int si = srcA[e], di = dstA[e];
    bool fb = (!cov[si] && s >= __int_as_float(bestBits[si])) ||
              (!cov[di] && s >= __int_as_float(bestBits[di]));
    keep[e] = keep[e] | (fb ? 1 : 0);
  }
  __syncthreads();

  // budget
  int nk = 0;
  for (int base = 0; base < E_; base += 32) {
    int e = base + lane, k = keep[e];
    unsigned m = (unsigned)__ballot(k != 0);
    if (k) k = (nk + __popc(m & ltmask)) < BUD_ ? 1 : 0;
    nk += __popc(m);
    keep[e] = k;
  }

  // pack (kept first, stable)
  int totK = 0;
  for (int base = 0; base < E_; base += 32)
    totK += __popc((unsigned)__ballot(keep[base + lane] != 0));
  if (lane == 0) atomicAdd(&accums[9], (float)totK);  // for kept_ratio

  float* o_idx  = out;
  float* o_rel  = out + (size_t)B_ * 2 * E_;
  float* o_mask = o_rel + (size_t)B_ * E_;
  float* o_scr  = o_mask + (size_t)B_ * E_;
  int kc = 0, dc = 0;
  for (int base = 0; base < E_; base += 32) {
    int e = base + lane, k = keep[e];
    unsigned m = (unsigned)__ballot(k != 0);
    int bk = __popc(m & ltmask), nk32 = __popc(m);
    int bd = __popc(~m & ltmask);
    int pos = k ? (kc + bk) : (totK + dc + bd);
    o_idx[(size_t)b * 2 * E_ + pos]      = (float)srcA[e];
    o_idx[(size_t)b * 2 * E_ + E_ + pos] = (float)dstA[e];
    o_rel[(size_t)b * E_ + pos]  = k ? (float)rel[e] : 0.f;
    o_mask[(size_t)b * E_ + pos] = k ? 1.f : 0.f;
    o_scr[(size_t)b * E_ + pos]  = k ? sc[e] : 0.f;
    kc += nk32; dc += (32 - nk32);
  }
}

__global__ void k_zero_accums(float* accums) {
  if (threadIdx.x < 16) accums[threadIdx.x] = 0.f;
}

__global__ void k_finalize(const float* __restrict__ accums, float* __restrict__ out) {
  const float inv = 1.0f / (float)(B_ * E_);
  const size_t o_load = (size_t)B_ * 2 * E_ + 3 * (size_t)B_ * E_;
  int t = threadIdx.x;
  if (t < NEXP_) out[o_load + t] = accums[t] * inv;                     // load
  if (t == NEXP_)     out[o_load + NEXP_]     = accums[9] * inv;        // kept_ratio
  if (t == NEXP_ + 1) out[o_load + NEXP_ + 1] = 1.0f - accums[8] * inv; // overflow
}

// ---------------------------------------------------------------------------
// launch
// ---------------------------------------------------------------------------
extern "C" void kernel_launch(void* const* d_in, const int* in_sizes, int n_in,
                              void* d_out, int out_size, void* d_ws, size_t ws_size,
                              hipStream_t stream) {
  (void)in_sizes; (void)n_in; (void)out_size; (void)ws_size;
  const float* node_x     = (const float*)d_in[0];
  const int*   edge_index = (const int*)d_in[1];
  const float* edge_bank  = (const float*)d_in[2];
  const float* community  = (const float*)d_in[3];
  const float* W_node = (const float*)d_in[4];  const float* b_node = (const float*)d_in[5];
  const float* W_edge = (const float*)d_in[6];  const float* b_edge = (const float*)d_in[7];
  const float* W_comm = (const float*)d_in[8];  const float* b_comm = (const float*)d_in[9];
  const float* ln_c_g = (const float*)d_in[10]; const float* ln_c_b = (const float*)d_in[11];
  const float* W_clu  = (const float*)d_in[12]; const float* b_clu  = (const float*)d_in[13];
  const float* W_tok  = (const float*)d_in[14]; const float* b_tok  = (const float*)d_in[15];
  const float* W_r1   = (const float*)d_in[16]; const float* b_r1   = (const float*)d_in[17];
  const float* ln_r_g = (const float*)d_in[18]; const float* ln_r_b = (const float*)d_in[19];
  const float* W_r2   = (const float*)d_in[20]; const float* b_r2   = (const float*)d_in[21];
  const float* head_prompts = (const float*)d_in[22];
  float* out = (float*)d_out;

  // workspace carve (4-byte units)
  unsigned* wsU = (unsigned*)d_ws;
  size_t o = 0;
  auto carve = [&](size_t n) { unsigned* p = wsU + o; o += n; return p; };
  unsigned* WnodePk = carve(128 * 256);
  unsigned* WedgePk = carve(16 * 256);
  unsigned* WcommPk = carve(16 * 256);
  unsigned* WcluPk  = carve(128 * 256);
  unsigned* WtokPk  = carve(4 * 128 * 256);
  unsigned* Wr1Pk   = carve(128 * 256);
  unsigned* Wr2Pk   = carve(128 * 256);
  float*    node_feat = (float*)carve((size_t)B_ * R_ * 256);
  unsigned* node_bfpk = carve((size_t)B_ * R_ * 128);
  float*    comm      = (float*)carve((size_t)B_ * R_ * 256);
  int*      relW      = (int*)carve((size_t)B_ * E_);
  float*    scoreW    = (float*)carve((size_t)B_ * E_);
  int*      keepWS    = (int*)carve((size_t)B_ * E_);
  float*    accums    = (float*)carve(16);

  // 1) pack weights into fragment-packed bf16 layout
  k_pack_weight_frag<<<128, 256, 0, stream>>>(W_node, WnodePk, 256, 8);
  k_pack_weight_frag<<<16,  256, 0, stream>>>(W_edge, WedgePk, 16,  1);
  k_pack_weight_frag<<<16,  256, 0, stream>>>(W_comm, WcommPk, 32,  1);
  k_pack_weight_frag<<<128, 256, 0, stream>>>(W_clu,  WcluPk,  256, 8);
  for (int p = 0; p < 4; p++)
    k_pack_weight_frag<<<128, 256, 0, stream>>>(W_tok + (size_t)p * 256 * 256,
                                                WtokPk + (size_t)p * 128 * 256, 256, 8);
  k_pack_weight_frag<<<128, 256, 0, stream>>>(W_r1, Wr1Pk, 256, 8);
  k_pack_weight_frag<<<128, 256, 0, stream>>>(W_r2, Wr2Pk, 256, 8);

  // 2) node_feat / comm_feat GEMMs (B*R = 1024 rows)
  k_gemm_rows<<<64, 256, 0, stream>>>(node_x, 256, WnodePk, b_node, node_feat);
  k_gemm_rows<<<64, 256, 0, stream>>>(community, 32, WcommPk, b_comm, comm);
  k_pack_act<<<(B_ * R_ * 128 + 255) / 256, 256, 0, stream>>>(
      node_feat, node_bfpk, B_ * R_ * 128);

  // 3) fused per-edge pipeline (dynamic LDS ~225KB; CDNA5 WGP has 320KB)
  k_zero_accums<<<1, 32, 0, stream>>>(accums);
  k_edges<<<B_ * (E_ / ET_), 256, SMEM_WORDS_ * 4, stream>>>(
      edge_bank, edge_index, node_bfpk, comm,
      WedgePk, b_edge, ln_c_g, ln_c_b, WcluPk, b_clu,
      WtokPk, b_tok, Wr1Pk, b_r1, ln_r_g, ln_r_b, Wr2Pk, b_r2,
      head_prompts, relW, scoreW, accums);

  // 4) capacity/guard/budget/pack + outputs
  k_select_pack<<<B_, 32, 0, stream>>>(edge_index, relW, scoreW, keepWS, out, accums);
  k_finalize<<<1, 32, 0, stream>>>(accums, out);
}
